// RoPEMultiHeadAttention_54073638256669
// MI455X (gfx1250) — matrix operand order repaired and checked by reading, not compiled, not run on hardware
//
#include <hip/hip_runtime.h>

typedef _Float16 half_t;
typedef __attribute__((ext_vector_type(16))) _Float16 v16h;
typedef __attribute__((ext_vector_type(8)))  float    v8f;
typedef int b128_t __attribute__((vector_size(16)));   // async-LDS transfer unit

#define B_   128
#define S_   64
#define D_   2048
#define H_   16
#define HD_  128
#define MTOT (B_ * S_)   // 8192

// CDNA5 async global->LDS staging (ASYNCcnt path), with safe fallback.
#if defined(__AMDGCN__) && \
    __has_builtin(__builtin_amdgcn_global_load_async_to_lds_b128) && \
    __has_builtin(__builtin_amdgcn_s_wait_asynccnt)
#define USE_ASYNC_LDS 1
#define GPTR(p) ((__attribute__((address_space(1))) b128_t*)(p))
#define LPTR(p) ((__attribute__((address_space(3))) b128_t*)(p))
#else
#define USE_ASYNC_LDS 0
#endif

union FragAB { v16h v; uint4 q[2]; };

enum { MODE_PLAIN = 0, MODE_ROPE = 1, MODE_VT = 2 };

// ---------------------------------------------------------------------------
// fp32 -> f16 conversion
// ---------------------------------------------------------------------------
__global__ void convert_f32_f16(const float* __restrict__ in,
                                half_t* __restrict__ out, int n4) {
    int i = blockIdx.x * blockDim.x + threadIdx.x;
    if (i < n4) {
        float4 f = ((const float4*)in)[i];
        union { half_t h[4]; uint2 u; } p;
        p.h[0] = (half_t)f.x; p.h[1] = (half_t)f.y;
        p.h[2] = (half_t)f.z; p.h[3] = (half_t)f.w;
        ((uint2*)out)[i] = p.u;
    }
}

// ---------------------------------------------------------------------------
// Tiled WMMA GEMM:  C[m][n] = sum_k A[m][k] * W[n][k] + bias[n]
// Block tile 128x128, BK=32, 8 waves each computing 64x32 (4x2 16x16 tiles).
// Double-buffered LDS; async global->LDS staging on CDNA5.
// ---------------------------------------------------------------------------
template <int MODE>
__global__ __launch_bounds__(256)
void gemm_wmma(const half_t* __restrict__ A,     // [MTOT][D] row-major f16
               const half_t* __restrict__ W,     // [D][D]    row-major f16 (n-major)
               const float*  __restrict__ bias,  // [D]
               void* __restrict__ out) {
    constexpr int K = D_, N = D_;
    constexpr int BM = 128, BN = 128, BK = 32;
    constexpr int LDR = BK + 8;                  // 40 halves: conflict-free stride
    __shared__ alignas(16) half_t As[2][BM][LDR];
    __shared__ alignas(16) half_t Bs[2][BN][LDR];

    const int tid  = threadIdx.x;
    const int lane = tid & 31;
    const int wave = tid >> 5;
    const int ln15 = lane & 15;
    const int hi   = lane >> 4;

    const int blockN = blockIdx.x * BN;
    const int blockM = blockIdx.y * BM;
    const int wm = (wave >> 2) * 64;             // 0 / 64
    const int wn = (wave & 3)  * 32;             // 0 / 32 / 64 / 96

    // cooperative global->LDS mapping: thread stages 16 halves of A and of B
    const int grow = tid >> 1;                   // 0..127
    const int gcol = (tid & 1) << 4;             // 0 or 16 halves
    const half_t* Ag = A + (size_t)(blockM + grow) * K + gcol;
    const half_t* Bg = W + (size_t)(blockN + grow) * K + gcol;

    v8f acc[4][2] = {};

#if USE_ASYNC_LDS
#define STAGE_TILE(nb, kofs)                                                   \
    do {                                                                       \
        __builtin_amdgcn_global_load_async_to_lds_b128(                        \
            GPTR(Ag + (kofs)),      LPTR(&As[nb][grow][gcol]),     0, 0);      \
        __builtin_amdgcn_global_load_async_to_lds_b128(                        \
            GPTR(Ag + (kofs) + 8),  LPTR(&As[nb][grow][gcol + 8]), 0, 0);      \
        __builtin_amdgcn_global_load_async_to_lds_b128(                        \
            GPTR(Bg + (kofs)),      LPTR(&Bs[nb][grow][gcol]),     0, 0);      \
        __builtin_amdgcn_global_load_async_to_lds_b128(                        \
            GPTR(Bg + (kofs) + 8),  LPTR(&Bs[nb][grow][gcol + 8]), 0, 0);      \
    } while (0)

    STAGE_TILE(0, 0);
    __builtin_amdgcn_s_wait_asynccnt(0);
#else
    uint4 a0 = *(const uint4*)(Ag);
    uint4 a1 = *(const uint4*)(Ag + 8);
    uint4 b0 = *(const uint4*)(Bg);
    uint4 b1 = *(const uint4*)(Bg + 8);
    *(uint4*)&As[0][grow][gcol]     = a0;
    *(uint4*)&As[0][grow][gcol + 8] = a1;
    *(uint4*)&Bs[0][grow][gcol]     = b0;
    *(uint4*)&Bs[0][grow][gcol + 8] = b1;
#endif
    __syncthreads();

    constexpr int KSTEPS = K / BK;               // 64
    int buf = 0;
    for (int ks = 0; ks < KSTEPS; ++ks) {
        const bool have_next = (ks + 1 < KSTEPS);
#if USE_ASYNC_LDS
        if (have_next) STAGE_TILE(buf ^ 1, (ks + 1) * BK);  // in-flight copy
#else
        if (have_next) {                          // prefetch next tile to regs
            const half_t* An = Ag + (ks + 1) * BK;
            const half_t* Bn = Bg + (ks + 1) * BK;
            a0 = *(const uint4*)(An);
            a1 = *(const uint4*)(An + 8);
            b0 = *(const uint4*)(Bn);
            b1 = *(const uint4*)(Bn + 8);
        }
#endif
        // A frag: lane m=ln15, halves k = {8hi..8hi+7, 16+8hi..23+8hi}
        FragAB af[4], bf[2];
#pragma unroll
        for (int mt = 0; mt < 4; ++mt) {
            const half_t* r = &As[buf][wm + mt * 16 + ln15][0];
            af[mt].q[0] = *(const uint4*)(r + 8 * hi);
            af[mt].q[1] = *(const uint4*)(r + 16 + 8 * hi);
        }
        // B frag: lane n=ln15, halves k = {8hi..8hi+7, 16+8hi..23+8hi}
#pragma unroll
        for (int nt = 0; nt < 2; ++nt) {
            const half_t* r = &Bs[buf][wn + nt * 16 + ln15][0];
            bf[nt].q[0] = *(const uint4*)(r + 8 * hi);
            bf[nt].q[1] = *(const uint4*)(r + 16 + 8 * hi);
        }
#pragma unroll
        for (int mt = 0; mt < 4; ++mt)
#pragma unroll
            for (int nt = 0; nt < 2; ++nt)
                acc[mt][nt] = __builtin_amdgcn_wmma_f32_16x16x32_f16(
                    false, af[mt].v, false, bf[nt].v, (short)0, acc[mt][nt],
                    false, false);

        if (have_next) {
#if USE_ASYNC_LDS
            __builtin_amdgcn_s_wait_asynccnt(0);  // our LDS writes landed
#else
            int nb = buf ^ 1;
            *(uint4*)&As[nb][grow][gcol]     = a0;
            *(uint4*)&As[nb][grow][gcol + 8] = a1;
            *(uint4*)&Bs[nb][grow][gcol]     = b0;
            *(uint4*)&Bs[nb][grow][gcol + 8] = b1;
#endif
        }
        __syncthreads();
        buf ^= 1;
    }
#if USE_ASYNC_LDS
#undef STAGE_TILE
#endif

    // ---------------- epilogue ----------------
#pragma unroll
    for (int mt = 0; mt < 4; ++mt) {
#pragma unroll
        for (int nt = 0; nt < 2; ++nt) {
            const int n  = blockN + wn + nt * 16 + ln15;
            const float bn = bias[n];
            if (MODE == MODE_PLAIN) {
                float* o = (float*)out;
#pragma unroll
                for (int j = 0; j < 8; ++j) {
                    int m = blockM + wm + mt * 16 + 8 * hi + j;
                    o[(size_t)m * N + n] = acc[mt][nt][j] + bn;
                }
            } else if (MODE == MODE_ROPE) {
                half_t* o = (half_t*)out;
                const int h = n >> 7, hd = n & (HD_ - 1);
                const float freq =
                    __expf(-(float)(hd & ~1) * (9.210340371976184f / (float)HD_));
#pragma unroll
                for (int j = 0; j < 8; ++j) {
                    int m = blockM + wm + mt * 16 + 8 * hi + j;
                    int s = m & (S_ - 1), b = m >> 6;
                    float v = acc[mt][nt][j] + bn;
                    float other = __shfl_xor(v, 1, 32);   // paired channel
                    float ang = (float)s * freq;
                    float sn = __sinf(ang), cs = __cosf(ang);
                    float res = ((n & 1) == 0) ? (v * cs - other * sn)
                                               : (other * sn + v * cs);
                    o[(((size_t)b * H_ + h) * S_ + s) * HD_ + hd] = (half_t)res;
                }
            } else {  // MODE_VT: store V transposed [b][h][hd][s]
                half_t* o = (half_t*)out;
                const int h = n >> 7, hd = n & (HD_ - 1);
#pragma unroll
                for (int j = 0; j < 8; ++j) {
                    int m = blockM + wm + mt * 16 + 8 * hi + j;
                    int s = m & (S_ - 1), b = m >> 6;
                    o[(((size_t)b * H_ + h) * HD_ + hd) * S_ + s] =
                        (half_t)(acc[mt][nt][j] + bn);
                }
            }
        }
    }
}

// ---------------------------------------------------------------------------
// Attention: one block per (b,h); 4 waves x 16 query rows.
// scores = Q K^T * scale -> softmax -> O = P V ;  everything WMMA f16.
// ---------------------------------------------------------------------------
__global__ __launch_bounds__(128)
void attention_wmma(const half_t* __restrict__ qh,   // [B][H][S][HD]
                    const half_t* __restrict__ kh,   // [B][H][S][HD]
                    const half_t* __restrict__ vt,   // [B][H][HD][S]
                    half_t* __restrict__ ctx) {      // [B][S][D]
    __shared__ alignas(16) half_t sp[4][16][72];     // per-wave P tile (16x64)

    const int bh   = blockIdx.x;
    const int b    = bh >> 4, h = bh & (H_ - 1);
    const int tid  = threadIdx.x;
    const int lane = tid & 31;
    const int wave = tid >> 5;
    const int ln15 = lane & 15;
    const int hi   = lane >> 4;
    const int qbase = wave * 16;

    // Q as A-fragments: 4 k-steps over HD=128
    const half_t* qrow = qh + ((size_t)bh * S_ + qbase + ln15) * HD_;
    FragAB qa[4];
#pragma unroll
    for (int ks = 0; ks < 4; ++ks) {
        qa[ks].q[0] = *(const uint4*)(qrow + ks * 32 + 8 * hi);
        qa[ks].q[1] = *(const uint4*)(qrow + ks * 32 + 16 + 8 * hi);
    }

    // scores: 4 key tiles of 16
    v8f sc[4] = {};
    const half_t* krow = kh + ((size_t)bh * S_ + ln15) * HD_;
#pragma unroll
    for (int nt = 0; nt < 4; ++nt) {
        const half_t* kr = krow + (size_t)nt * 16 * HD_;
#pragma unroll
        for (int ks = 0; ks < 4; ++ks) {
            FragAB kb;
            kb.q[0] = *(const uint4*)(kr + ks * 32 + 8 * hi);
            kb.q[1] = *(const uint4*)(kr + ks * 32 + 16 + 8 * hi);
            sc[nt] = __builtin_amdgcn_wmma_f32_16x16x32_f16(
                false, qa[ks].v, false, kb.v, (short)0, sc[nt], false, false);
        }
    }

    // softmax over 64 keys (row lives in a fixed 16-lane half)
    const float scale = 0.08838834764831845f;        // HD^-0.5
#pragma unroll
    for (int j = 0; j < 8; ++j) {
        float mx = -3.4e38f;
#pragma unroll
        for (int nt = 0; nt < 4; ++nt) mx = fmaxf(mx, sc[nt][j]);
#pragma unroll
        for (int d = 1; d < 16; d <<= 1) mx = fmaxf(mx, __shfl_xor(mx, d, 32));
        float e[4], sum = 0.f;
#pragma unroll
        for (int nt = 0; nt < 4; ++nt) {
            e[nt] = __expf((sc[nt][j] - mx) * scale);
            sum += e[nt];
        }
#pragma unroll
        for (int d = 1; d < 16; d <<= 1) sum += __shfl_xor(sum, d, 32);
        float inv = 1.f / sum;
#pragma unroll
        for (int nt = 0; nt < 4; ++nt) sc[nt][j] = e[nt] * inv;
    }

    // transpose P through LDS into A-fragment layout
#pragma unroll
    for (int nt = 0; nt < 4; ++nt)
#pragma unroll
        for (int j = 0; j < 8; ++j)
            sp[wave][8 * hi + j][nt * 16 + ln15] = (half_t)sc[nt][j];
    __syncthreads();

    FragAB pa[2];
    const half_t* prow = &sp[wave][ln15][0];
#pragma unroll
    for (int ks = 0; ks < 2; ++ks) {
        pa[ks].q[0] = *(const uint4*)(prow + ks * 32 + 8 * hi);
        pa[ks].q[1] = *(const uint4*)(prow + ks * 32 + 16 + 8 * hi);
    }

    // O = P @ V : 8 hd-tiles, K=64 (2 steps); vt rows are B-operand native
    const half_t* vrow = vt + ((size_t)bh * HD_ + ln15) * S_;
#pragma unroll
    for (int nt = 0; nt < 8; ++nt) {
        v8f oc = {};
        const half_t* vr = vrow + (size_t)nt * 16 * S_;
#pragma unroll
        for (int ks = 0; ks < 2; ++ks) {
            FragAB vb;
            vb.q[0] = *(const uint4*)(vr + ks * 32 + 8 * hi);
            vb.q[1] = *(const uint4*)(vr + ks * 32 + 16 + 8 * hi);
            oc = __builtin_amdgcn_wmma_f32_16x16x32_f16(
                false, pa[ks].v, false, vb.v, (short)0, oc, false, false);
        }
        const int hd = nt * 16 + ln15;
#pragma unroll
        for (int j = 0; j < 8; ++j) {
            int s = qbase + 8 * hi + j;
            ctx[((size_t)b * S_ + s) * D_ + h * HD_ + hd] = (half_t)oc[j];
        }
    }
}

// ---------------------------------------------------------------------------
extern "C" void kernel_launch(void* const* d_in, const int* in_sizes, int n_in,
                              void* d_out, int out_size, void* d_ws,
                              size_t ws_size, hipStream_t stream) {
    (void)in_sizes; (void)n_in; (void)out_size; (void)ws_size;
    const float* x  = (const float*)d_in[0];
    const float* Wq = (const float*)d_in[1];
    const float* bq = (const float*)d_in[2];
    const float* Wk = (const float*)d_in[3];
    const float* bk = (const float*)d_in[4];
    const float* Wv = (const float*)d_in[5];
    const float* bv = (const float*)d_in[6];
    const float* Wo = (const float*)d_in[7];
    const float* bo = (const float*)d_in[8];
    float* out = (float*)d_out;

    half_t* ws = (half_t*)d_ws;
    size_t off = 0;
    half_t* xh = ws + off; off += (size_t)MTOT * D_;
    half_t* wq = ws + off; off += (size_t)D_ * D_;
    half_t* wk = ws + off; off += (size_t)D_ * D_;
    half_t* wv = ws + off; off += (size_t)D_ * D_;
    half_t* wo = ws + off; off += (size_t)D_ * D_;
    half_t* q  = ws + off; off += (size_t)MTOT * D_;
    half_t* k  = ws + off; off += (size_t)MTOT * D_;
    half_t* vT = ws + off; off += (size_t)MTOT * D_;
    half_t* ctx = xh;  // xh is dead once projections finish

    auto cvt = [&](const float* src, half_t* dst, size_t n) {
        int n4 = (int)(n / 4);
        convert_f32_f16<<<(n4 + 255) / 256, 256, 0, stream>>>(src, dst, n4);
    };
    cvt(x,  xh, (size_t)MTOT * D_);
    cvt(Wq, wq, (size_t)D_ * D_);
    cvt(Wk, wk, (size_t)D_ * D_);
    cvt(Wv, wv, (size_t)D_ * D_);
    cvt(Wo, wo, (size_t)D_ * D_);

    dim3 gg(D_ / 128, MTOT / 128);
    gemm_wmma<MODE_ROPE><<<gg, 256, 0, stream>>>(xh, wq, bq, (void*)q);
    gemm_wmma<MODE_ROPE><<<gg, 256, 0, stream>>>(xh, wk, bk, (void*)k);
    gemm_wmma<MODE_VT>  <<<gg, 256, 0, stream>>>(xh, wv, bv, (void*)vT);

    attention_wmma<<<B_ * H_, 128, 0, stream>>>(q, k, vT, ctx);

    gemm_wmma<MODE_PLAIN><<<gg, 256, 0, stream>>>(ctx, wo, bo, (void*)out);
}
